// CustomMHA_33870112096740
// MI455X (gfx1250) — matrix operand order repaired and checked by reading, not compiled, not run on hardware
//
#include <hip/hip_runtime.h>

// ---------------------------------------------------------------------------
// CustomMHA for MI455X (gfx1250): bf16 WMMA everywhere, f32 accumulate.
// B=4, S=2048, D=1024, H=16, HD=64.  Compute-bound (~137 GFLOP vs ~300MB HBM).
// Weight tiles stream through the Tensor Data Mover into double-buffered LDS;
// the TDM issue is branchless (wave w issues matrix w's tile, wave 3 issues a
// NULL descriptor) because TDM ops ignore EXEC.
// ---------------------------------------------------------------------------

typedef __bf16 bf16;
typedef __attribute__((ext_vector_type(16))) __bf16 v16bf;
typedef __attribute__((ext_vector_type(8)))  __bf16 v8bf;
typedef __attribute__((ext_vector_type(8)))  float  v8f;
typedef __attribute__((ext_vector_type(4)))  unsigned u32x4;
typedef __attribute__((ext_vector_type(8)))  unsigned u32x8;

#define BQ 4
#define SQ 2048
#define DQ 1024
#define HQ 16
#define HDQ 64

union AFrag { v16bf v; v8bf h[2]; };

__device__ __forceinline__ v8f wmma_bf16(v16bf a, v16bf b, v8f c) {
  // D = A(16x32 bf16) * B(32x16 bf16) + C(16x16 f32)
  return __builtin_amdgcn_wmma_f32_16x16x32_bf16(false, a, false, b,
                                                 (short)0, c, false, false);
}

// TDM: load a 64(row) x 32(halves) bf16 tile (row stride 1024 halves) from
// global into LDS at ldsOff.  D# per CDNA5 ISA.
// count=0 -> NULL tensor descriptor (architectural no-op), used so every wave
// can execute the instruction branchlessly (TDM ignores EXEC).
__device__ __forceinline__ void tdm_load_tile(const bf16* gsrc, unsigned ldsOff,
                                              unsigned count) {
  const unsigned long long ga = (unsigned long long)(uintptr_t)gsrc;
  const u32x4 g0 = {
      count,                                           // count (0=NULL, 1=valid)
      ldsOff,                                          // lds_addr (bytes)
      (unsigned)ga,                                    // global_addr[31:0]
      (unsigned)((ga >> 32) & 0x1FFFFFFull) | (2u << 30)  // addr[56:32]|type=2
  };
  const u32x8 g1 = {
      1u << 16,           // wg_mask=0, data_size=1 (2 bytes)
      1024u << 16,        // tensor_dim0 = 1024 (lo16 @ bits 63:48)
      0u,                 // tensor_dim0 hi16 = 0, tensor_dim1 lo16 = 0
      16u | (32u << 16),  // tensor_dim1 hi16 (=1<<20) | tile_dim0 = 32
      64u,                // tile_dim1 = 64, tile_dim2 = 0
      1024u,              // tensor_dim0_stride = 1024 (lo32)
      0u, 0u };           // stride0 hi, stride1 (unused for 2D tile)
  asm volatile("tensor_load_to_lds %0, %1" :: "s"(g0), "s"(g1) : "memory");
}

// ---------------------------------------------------------------------------
__global__ void f32_to_bf16_kernel(const float* __restrict__ in,
                                   bf16* __restrict__ out, int n) {
  int i = blockIdx.x * blockDim.x + threadIdx.x;
  if (i < n) out[i] = (bf16)in[i];
}

// ---------------------------------------------------------------------------
// Fused Q/K/V projection + RoPE.  One wave owns a 16(M) x 64(N = one head)
// output tile for all three matrices.  The 4 waves of a block share one head:
// per K-step, wave w (w<3) TDM-loads weight-matrix w's 64x32 tile into LDS
// (double buffered, 24KB/block); everyone reads B-fragments from LDS.
__global__ __launch_bounds__(128) void qkv_rope_kernel(
    const bf16* __restrict__ xb,
    const bf16* __restrict__ wq, const bf16* __restrict__ wk,
    const bf16* __restrict__ wv,
    const float* __restrict__ bq, const float* __restrict__ bk,
    const float* __restrict__ bv,
    const float* __restrict__ cosT, const float* __restrict__ sinT,
    bf16* __restrict__ qo, bf16* __restrict__ ko, bf16* __restrict__ vto) {
  __shared__ __align__(64) bf16 wtile[2][3][64 * 32];

  const int wave = threadIdx.x >> 5;
  const int lane = threadIdx.x & 31;
  const int head   = blockIdx.x & (HQ - 1);
  const int mgroup = blockIdx.x >> 4;
  const int M0 = (mgroup * 4 + wave) * 16;   // 512 M-tiles total

  const int colsel = lane & 15;
  const int kk   = (lane < 16) ? 0 : 16;
  const int kb16 = (lane < 16) ? 0 : 16;
  const int mofs = (lane >= 16) ? 8 : 0;

  v8f cq[4], ck[4], cv[4];
#pragma unroll
  for (int t = 0; t < 4; ++t) { cq[t] = {}; ck[t] = {}; cv[t] = {}; }

  const bf16* aRow = xb + (size_t)(M0 + colsel) * DQ + kk;

  // branchless per-wave TDM issue: wave 0->Wq, 1->Wk, 2->Wv, 3->NULL desc
  const int mat = (wave < 3) ? wave : 0;
  const unsigned cnt = (wave < 3) ? 1u : 0u;
  const bf16* wsrc = (mat == 0) ? wq : ((mat == 1) ? wk : wv);
  const bf16* wsrcHead = wsrc + (size_t)head * HDQ * DQ;

  auto issue_tile = [&](int buf, int k0) {
    tdm_load_tile(wsrcHead + k0, (unsigned)(uintptr_t)&wtile[buf][mat][0], cnt);
  };

  issue_tile(0, 0);

  for (int k0 = 0; k0 < DQ; k0 += 32) {
    const int buf = (k0 >> 5) & 1;
    __builtin_amdgcn_s_wait_tensorcnt(0);   // own DMA done (all waves, uniform)
    __syncthreads();                        // tile(buf) visible to all waves
    if ((k0 + 32) < DQ) issue_tile(buf ^ 1, k0 + 32);  // uniform scalar branch

    __builtin_prefetch(aRow + k0 + 128, 0, 1);   // global_prefetch_b8
    AFrag a;
    a.h[0] = *(const v8bf*)(aRow + k0);
    a.h[1] = *(const v8bf*)(aRow + k0 + 8);

    // load all 12 B-fragments, then 12 WMMAs
    v16bf fq[4], fk[4], fv[4];
#pragma unroll
    for (int t = 0; t < 4; ++t) {
      const int rowoff = (t * 16 + colsel) * 32 + kb16;
      fq[t] = *(const v16bf*)&wtile[buf][0][rowoff];
      fk[t] = *(const v16bf*)&wtile[buf][1][rowoff];
      fv[t] = *(const v16bf*)&wtile[buf][2][rowoff];
    }
#pragma unroll
    for (int t = 0; t < 4; ++t) {
      cq[t] = wmma_bf16(a.v, fq[t], cq[t]);
      ck[t] = wmma_bf16(a.v, fk[t], ck[t]);
      cv[t] = wmma_bf16(a.v, fv[t], cv[t]);
    }
    __syncthreads();                        // everyone done with tile(buf)
  }

  // Epilogue: bias + RoPE (partner hd+-32 is accumulator tile t^2, same lane/r)
#pragma unroll
  for (int r = 0; r < 8; ++r) {
    const int m = M0 + r + mofs;
    const int s = m & (SQ - 1);
    const int b = m >> 11;                 // S = 2048
    float qv[4], kv[4], vv[4], cs[4], sn[4];
#pragma unroll
    for (int t = 0; t < 4; ++t) {
      const int hd = t * 16 + colsel;
      const int n  = head * HDQ + hd;
      qv[t] = cq[t][r] + bq[n];
      kv[t] = ck[t][r] + bk[n];
      vv[t] = cv[t][r] + bv[n];
      cs[t] = cosT[s * HDQ + hd];
      sn[t] = sinT[s * HDQ + hd];
    }
    const size_t qkRow = ((size_t)(b * HQ + head) * SQ + s) * HDQ;
#pragma unroll
    for (int t = 0; t < 2; ++t) {
      const int hd0 = t * 16 + colsel;
      // first half: x1*cos - x2*sin ; second half: x2*cos + x1*sin
      float q0 = qv[t] * cs[t]         - qv[t + 2] * sn[t];
      float q1 = qv[t + 2] * cs[t + 2] + qv[t] * sn[t + 2];
      float k0v = kv[t] * cs[t]         - kv[t + 2] * sn[t];
      float k1v = kv[t + 2] * cs[t + 2] + kv[t] * sn[t + 2];
      qo[qkRow + hd0]      = (bf16)q0;
      qo[qkRow + hd0 + 32] = (bf16)q1;
      ko[qkRow + hd0]      = (bf16)k0v;
      ko[qkRow + hd0 + 32] = (bf16)k1v;
    }
#pragma unroll
    for (int t = 0; t < 4; ++t) {          // V stored transposed: [B,H,hd,S]
      const int hd = t * 16 + colsel;
      vto[((size_t)(b * HQ + head) * HDQ + hd) * SQ + s] = (bf16)vv[t];
    }
  }
}

// ---------------------------------------------------------------------------
// Causal flash attention: one wave per (b, h, 16-query tile); online softmax
// over 32-key blocks; P routed through per-wave LDS (C-layout -> A-layout).
// V fragments are loaded before the softmax so their latency overlaps it.
__global__ __launch_bounds__(128) void flash_attn_kernel(
    const bf16* __restrict__ qbuf, const bf16* __restrict__ kbuf,
    const bf16* __restrict__ vtbuf, bf16* __restrict__ ctx) {
  __shared__ __align__(32) bf16 plds[4][16 * 32];

  const int wave = threadIdx.x >> 5;
  const int lane = threadIdx.x & 31;
  const int bh     = blockIdx.x >> 5;            // 64 (b,h) pairs
  const int qgroup = blockIdx.x & 31;
  const int qbase  = (qgroup * 4 + wave) * 16;   // 128 q-tiles

  const int colsel = lane & 15;
  const int kk   = (lane < 16) ? 0 : 16;
  const int kb16 = (lane < 16) ? 0 : 16;
  const int mofs = (lane >= 16) ? 8 : 0;

  const size_t seqBase = (size_t)bh * SQ;
  const bf16* vbase = vtbuf + (size_t)bh * HDQ * SQ;

  // Q fragments (resident whole kernel): hd 0..31 and 32..63
  const bf16* qp = qbuf + (seqBase + qbase + colsel) * HDQ + kk;
  AFrag aq0, aq1;
  aq0.h[0] = *(const v8bf*)(qp);       aq0.h[1] = *(const v8bf*)(qp + 8);
  aq1.h[0] = *(const v8bf*)(qp + 32);  aq1.h[1] = *(const v8bf*)(qp + 40);

  float mi[8], li[8];
  v8f o[4];
#pragma unroll
  for (int r = 0; r < 8; ++r) { mi[r] = -1e30f; li[r] = 0.0f; }
#pragma unroll
  for (int t = 0; t < 4; ++t) o[t] = {};

  const int nkb = (qbase + 15) / 32 + 1;         // causal: keys <= q only
  for (int kb = 0; kb < nkb; ++kb) {
    const int kbase = kb * 32;
    // K fragments for both 16-key subtiles (distinct regs)
    const bf16* kp = kbuf + (seqBase + kbase + colsel) * HDQ;
    v16bf fk00 = *(const v16bf*)(kp + kb16);
    v16bf fk01 = *(const v16bf*)(kp + 32 + kb16);
    v16bf fk10 = *(const v16bf*)(kp + 16 * HDQ + kb16);
    v16bf fk11 = *(const v16bf*)(kp + 16 * HDQ + 32 + kb16);
    // V fragments early: independent of P, latency hidden under softmax
    v16bf fv[4];
#pragma unroll
    for (int t = 0; t < 4; ++t)
      fv[t] = *(const v16bf*)(vbase + (size_t)(t * 16 + colsel) * SQ
                              + kbase + kb16);
    v8f z = {};
    v8f sa0 = wmma_bf16(aq0.v, fk00, z);
    sa0 = wmma_bf16(aq1.v, fk01, sa0);
    v8f sa1 = wmma_bf16(aq0.v, fk10, z);
    sa1 = wmma_bf16(aq1.v, fk11, sa1);

    v8f st[2];
    const int kp0 = kbase + colsel;
#pragma unroll
    for (int r = 0; r < 8; ++r) {
      const int q = qbase + r + mofs;
      st[0][r] = (kp0 > q)      ? -1e30f : sa0[r] * 0.125f;  // 1/sqrt(HD)
      st[1][r] = (kp0 + 16 > q) ? -1e30f : sa1[r] * 0.125f;
    }
    // online softmax: row stats live per (r, 16-lane half)
#pragma unroll
    for (int r = 0; r < 8; ++r) {
      float mx = fmaxf(st[0][r], st[1][r]);
      mx = fmaxf(mx, __shfl_xor(mx, 1, 32));
      mx = fmaxf(mx, __shfl_xor(mx, 2, 32));
      mx = fmaxf(mx, __shfl_xor(mx, 4, 32));
      mx = fmaxf(mx, __shfl_xor(mx, 8, 32));
      const float mnew = fmaxf(mi[r], mx);
      const float alpha = expf(mi[r] - mnew);
      mi[r] = mnew;
      const float p0 = expf(st[0][r] - mnew);
      const float p1 = expf(st[1][r] - mnew);
      float rs = p0 + p1;
      rs += __shfl_xor(rs, 1, 32);
      rs += __shfl_xor(rs, 2, 32);
      rs += __shfl_xor(rs, 4, 32);
      rs += __shfl_xor(rs, 8, 32);
      li[r] = li[r] * alpha + rs;
#pragma unroll
      for (int t = 0; t < 4; ++t) o[t][r] *= alpha;
      const int m = r + mofs;                    // C-layout -> LDS row-major
      plds[wave][m * 32 + colsel]      = (bf16)p0;
      plds[wave][m * 32 + 16 + colsel] = (bf16)p1;
    }
    asm volatile("s_wait_dscnt 0" ::: "memory"); // LDS RAW fence (CDNA5)

    // reload P as A-fragment (16 q-rows x 32 keys)
    const bf16* pp = &plds[wave][colsel * 32 + kk];
    AFrag pa;
    pa.h[0] = *(const v8bf*)(pp);
    pa.h[1] = *(const v8bf*)(pp + 8);
#pragma unroll
    for (int t = 0; t < 4; ++t)                  // P(16x32) x V(32x16 hd)
      o[t] = wmma_bf16(pa.v, fv[t], o[t]);
  }

  // normalize + write ctx back in [B,S,D] layout for the output GEMM
  const int b = bh >> 4, h = bh & 15;
#pragma unroll
  for (int r = 0; r < 8; ++r) {
    const float inv = 1.0f / li[r];
    const int m = qbase + r + mofs;
    const size_t orow = ((size_t)b * SQ + m) * DQ + h * HDQ;
#pragma unroll
    for (int t = 0; t < 4; ++t)
      ctx[orow + t * 16 + colsel] = (bf16)(o[t][r] * inv);
  }
}

// ---------------------------------------------------------------------------
// Output projection: out = ctx @ Wo^T + bo   (fp32 output)
__global__ __launch_bounds__(128) void out_proj_kernel(
    const bf16* __restrict__ ctx, const bf16* __restrict__ wo,
    const float* __restrict__ bo, float* __restrict__ out) {
  const int wave = threadIdx.x >> 5;
  const int lane = threadIdx.x & 31;
  const int ncol   = blockIdx.x & 15;          // 16 column groups of 64
  const int mgroup = blockIdx.x >> 4;
  const int M0 = (mgroup * 4 + wave) * 16;

  const int colsel = lane & 15;
  const int kk   = (lane < 16) ? 0 : 16;
  const int kb16 = (lane < 16) ? 0 : 16;
  const int mofs = (lane >= 16) ? 8 : 0;

  v8f c[4];
#pragma unroll
  for (int t = 0; t < 4; ++t) c[t] = {};

  const bf16* aRow = ctx + (size_t)(M0 + colsel) * DQ + kk;
  const bf16* bRow = wo + (size_t)(ncol * HDQ + colsel) * DQ + kb16;
  for (int k0 = 0; k0 < DQ; k0 += 32) {
    __builtin_prefetch(aRow + k0 + 128, 0, 1);
    AFrag a;
    a.h[0] = *(const v8bf*)(aRow + k0);
    a.h[1] = *(const v8bf*)(aRow + k0 + 8);
    v16bf f[4];
#pragma unroll
    for (int t = 0; t < 4; ++t)
      f[t] = *(const v16bf*)(bRow + (size_t)t * 16 * DQ + k0);
#pragma unroll
    for (int t = 0; t < 4; ++t)
      c[t] = wmma_bf16(a.v, f[t], c[t]);
  }
#pragma unroll
  for (int r = 0; r < 8; ++r) {
    const size_t row = (size_t)(M0 + r + mofs) * DQ;
#pragma unroll
    for (int t = 0; t < 4; ++t) {
      const int n = ncol * HDQ + t * 16 + colsel;
      out[row + n] = c[t][r] + bo[n];
    }
  }
}

// ---------------------------------------------------------------------------
extern "C" void kernel_launch(void* const* d_in, const int* in_sizes, int n_in,
                              void* d_out, int out_size, void* d_ws,
                              size_t ws_size, hipStream_t stream) {
  (void)in_sizes; (void)n_in; (void)out_size; (void)ws_size;
  const float* x    = (const float*)d_in[0];
  // d_in[1] = mask (causal, applied analytically -> unused)
  const float* rcos = (const float*)d_in[2];
  const float* rsin = (const float*)d_in[3];
  const float* Wq = (const float*)d_in[4];  const float* bq = (const float*)d_in[5];
  const float* Wk = (const float*)d_in[6];  const float* bk = (const float*)d_in[7];
  const float* Wv = (const float*)d_in[8];  const float* bv = (const float*)d_in[9];
  const float* Wo = (const float*)d_in[10]; const float* bo = (const float*)d_in[11];

  const int NX = BQ * SQ * DQ;      // 8388608
  const int NW = DQ * DQ;           // 1048576

  char* ws = (char*)d_ws;
  bf16* xb   = (bf16*)(ws);
  bf16* wqb  = (bf16*)(ws + 16u * 1024 * 1024);
  bf16* wkb  = (bf16*)(ws + 18u * 1024 * 1024);
  bf16* wvb  = (bf16*)(ws + 20u * 1024 * 1024);
  bf16* wob  = (bf16*)(ws + 22u * 1024 * 1024);
  bf16* qbuf = (bf16*)(ws + 24u * 1024 * 1024);
  bf16* kbuf = (bf16*)(ws + 40u * 1024 * 1024);
  bf16* vtb  = (bf16*)(ws + 56u * 1024 * 1024);
  bf16* ctxb = (bf16*)(ws + 72u * 1024 * 1024);   // total 88 MB

  f32_to_bf16_kernel<<<(NX + 255) / 256, 256, 0, stream>>>(x,  xb,  NX);
  f32_to_bf16_kernel<<<(NW + 255) / 256, 256, 0, stream>>>(Wq, wqb, NW);
  f32_to_bf16_kernel<<<(NW + 255) / 256, 256, 0, stream>>>(Wk, wkb, NW);
  f32_to_bf16_kernel<<<(NW + 255) / 256, 256, 0, stream>>>(Wv, wvb, NW);
  f32_to_bf16_kernel<<<(NW + 255) / 256, 256, 0, stream>>>(Wo, wob, NW);

  // 512 M-tiles x 16 heads, 4 waves/block -> 2048 blocks
  qkv_rope_kernel<<<2048, 128, 0, stream>>>(xb, wqb, wkb, wvb, bq, bk, bv,
                                            rcos, rsin, qbuf, kbuf, vtb);
  // 64 (b,h) x 32 q-groups -> 2048 blocks
  flash_attn_kernel<<<2048, 128, 0, stream>>>(qbuf, kbuf, vtb, ctxb);
  // 512 M-tiles x 16 col-groups, 4 waves/block -> 2048 blocks
  out_proj_kernel<<<2048, 128, 0, stream>>>(ctxb, wob, bo, (float*)d_out);
}
